// SlotAttention_8486855377213
// MI455X (gfx1250) — compile-verified, matrix operand order checked by reading, not run on hardware
//
#include <hip/hip_runtime.h>

// ---------------- problem constants ----------------
#define B_ 32
#define N_ 1024
#define D_ 1024
#define S_ 64
#define H_ 4096

// ---------------- CDNA5 WMMA types ----------------
typedef __attribute__((ext_vector_type(16))) __bf16 v16bf;
typedef __attribute__((ext_vector_type(8)))  float  v8f;

union Frag16 { v16bf v; unsigned int u[8]; };

__device__ __forceinline__ unsigned short f2bf(float x) {
  unsigned int u = __float_as_uint(x);
  u += 0x7FFFu + ((u >> 16) & 1u);   // round-to-nearest-even
  return (unsigned short)(u >> 16);
}

__device__ __forceinline__ float gelu_tanh(float x) {
  float c = 0.7978845608028654f * (x + 0.044715f * x * x * x);
  return 0.5f * x * (1.0f + tanhf(c));
}

// ---------------- CDNA5 async global->LDS copy (ASYNCcnt path) ----------------
#if defined(__has_builtin)
# if __has_builtin(__builtin_amdgcn_global_load_async_to_lds_b128)
#  define SA_ASYNC 1
# endif
#endif
#ifndef SA_ASYNC
# define SA_ASYNC 0
#endif

#if SA_ASYNC
typedef int sa_v4i __attribute__((ext_vector_type(4)));
typedef __attribute__((address_space(1))) sa_v4i* sa_gvec;   // global, v4i, non-const
typedef __attribute__((address_space(3))) sa_v4i* sa_lvec;   // LDS, v4i
__device__ __forceinline__ void sa_cp16_async(const void* g, void* l) {
  __builtin_amdgcn_global_load_async_to_lds_b128(
      (sa_gvec)(sa_v4i*)g, (sa_lvec)(sa_v4i*)l, 0, 0);
}
__device__ __forceinline__ void sa_async_wait0() {
# if __has_builtin(__builtin_amdgcn_s_wait_asynccnt)
  __builtin_amdgcn_s_wait_asynccnt(0);
# else
  asm volatile("s_wait_asynccnt 0x0" ::: "memory");
# endif
}
#endif

// ---------------- f32 -> bf16 transposed convert (weights) ----------------
// x: [R, C] f32 row-major  ->  y: [C, R] bf16 row-major
__global__ __launch_bounds__(256) void sa_cvt_bf16_t(const float* __restrict__ x,
                                                     unsigned short* __restrict__ y,
                                                     int R, int C) {
  __shared__ unsigned short tile[32][33];
  int c0 = blockIdx.x * 32, r0 = blockIdx.y * 32;
  int tx = threadIdx.x & 31, ty = threadIdx.x >> 5;   // 32 x 8
#pragma unroll
  for (int i = 0; i < 32; i += 8)
    tile[ty + i][tx] = f2bf(x[(size_t)(r0 + ty + i) * C + c0 + tx]);
  __syncthreads();
#pragma unroll
  for (int i = 0; i < 32; i += 8)
    y[(size_t)(c0 + ty + i) * R + r0 + tx] = tile[tx][ty + i];
}

// ---------------- bf16 batched transpose (V -> V^T per batch) ----------------
// x: [z][N_][D_] -> y: [z][D_][N_]
__global__ __launch_bounds__(256) void sa_transpose_bf16(const unsigned short* __restrict__ x,
                                                         unsigned short* __restrict__ y) {
  __shared__ unsigned short tile[32][33];
  long long zb = (long long)blockIdx.z * N_ * D_;
  int c0 = blockIdx.x * 32, r0 = blockIdx.y * 32;
  int tx = threadIdx.x & 31, ty = threadIdx.x >> 5;
#pragma unroll
  for (int i = 0; i < 32; i += 8)
    tile[ty + i][tx] = x[zb + (long long)(r0 + ty + i) * D_ + c0 + tx];
  __syncthreads();
#pragma unroll
  for (int i = 0; i < 32; i += 8)
    y[zb + (long long)(c0 + ty + i) * N_ + r0 + tx] = tile[tx][ty + i];
}

// ---------------- flat f32 -> bf16 convert (slots before head GEMM) ----------------
__global__ __launch_bounds__(256) void sa_cvt_bf16(const float* __restrict__ x,
                                                   unsigned short* __restrict__ y) {
  long long i = (long long)blockIdx.x * 256 + threadIdx.x;   // over elems/4
  float4 v = ((const float4*)x)[i];
  uint2 o;
  o.x = (unsigned)f2bf(v.x) | ((unsigned)f2bf(v.y) << 16);
  o.y = (unsigned)f2bf(v.z) | ((unsigned)f2bf(v.w) << 16);
  ((uint2*)y)[i] = o;
}

// ---------------- LayerNorm (f32 in, bf16 out), D = 1024 ----------------
__global__ __launch_bounds__(256) void sa_ln_bf16(const float* __restrict__ x,
                                                  const float* __restrict__ w,
                                                  const float* __restrict__ b,
                                                  unsigned short* __restrict__ y) {
  long long row = blockIdx.x;
  const float* xr = x + row * D_;
  int j4 = threadIdx.x * 4;
  float4 v = *(const float4*)(xr + j4);
  float s  = v.x + v.y + v.z + v.w;
  float sq = v.x * v.x + v.y * v.y + v.z * v.z + v.w * v.w;
  __shared__ float rs[256], rq[256];
  rs[threadIdx.x] = s; rq[threadIdx.x] = sq;
  __syncthreads();
  for (int off = 128; off > 0; off >>= 1) {
    if (threadIdx.x < off) { rs[threadIdx.x] += rs[threadIdx.x + off];
                             rq[threadIdx.x] += rq[threadIdx.x + off]; }
    __syncthreads();
  }
  float mean = rs[0] * (1.f / D_);
  float var  = rq[0] * (1.f / D_) - mean * mean;
  float rstd = rsqrtf(var + 1e-5f);
  float4 wv = *(const float4*)(w + j4);
  float4 bv = *(const float4*)(b + j4);
  uint2 o;
  o.x = (unsigned)f2bf((v.x - mean) * rstd * wv.x + bv.x) |
        ((unsigned)f2bf((v.y - mean) * rstd * wv.y + bv.y) << 16);
  o.y = (unsigned)f2bf((v.z - mean) * rstd * wv.z + bv.z) |
        ((unsigned)f2bf((v.w - mean) * rstd * wv.w + bv.w) << 16);
  *(uint2*)&y[row * D_ + j4] = o;
}

// ---------------- slot init (broadcast emb) ----------------
__global__ __launch_bounds__(256) void sa_init_slots(const float* __restrict__ emb,
                                                     float* __restrict__ slots) {
  long long i = (long long)blockIdx.x * 256 + threadIdx.x;   // over B*S*D/4
  ((float4*)slots)[i] = ((const float4*)emb)[i & (S_ * D_ / 4 - 1)];
}

// ---------------- softmax over the SLOT axis (per key column) ----------------
__global__ __launch_bounds__(256) void sa_softmax_slots(float* __restrict__ dots) {
  int b = blockIdx.y;
  int j = blockIdx.x * 256 + threadIdx.x;
  float* base = dots + (long long)b * S_ * N_ + j;
  float r[S_];
  float m = -3.4e38f;
#pragma unroll
  for (int i = 0; i < S_; ++i) { r[i] = base[(long long)i * N_]; m = fmaxf(m, r[i]); }
  float s = 0.f;
#pragma unroll
  for (int i = 0; i < S_; ++i) { r[i] = __expf(r[i] - m); s += r[i]; }
  float inv = 1.f / s;
#pragma unroll
  for (int i = 0; i < S_; ++i) base[(long long)i * N_] = r[i] * inv + 1e-4f;
}

// ---------------- renormalize over keys, emit bf16 attn ----------------
__global__ __launch_bounds__(256) void sa_renorm(const float* __restrict__ attn,
                                                 unsigned short* __restrict__ out) {
  long long row = (long long)blockIdx.y * S_ + blockIdx.x;   // (b, slot)
  const float* ar = attn + row * N_;
  int j4 = threadIdx.x * 4;
  float4 v = *(const float4*)(ar + j4);
  __shared__ float rs[256];
  rs[threadIdx.x] = v.x + v.y + v.z + v.w;
  __syncthreads();
  for (int off = 128; off > 0; off >>= 1) {
    if (threadIdx.x < off) rs[threadIdx.x] += rs[threadIdx.x + off];
    __syncthreads();
  }
  float inv = 1.f / rs[0];
  uint2 o;
  o.x = (unsigned)f2bf(v.x * inv) | ((unsigned)f2bf(v.y * inv) << 16);
  o.y = (unsigned)f2bf(v.z * inv) | ((unsigned)f2bf(v.w * inv) << 16);
  *(uint2*)&out[row * N_ + j4] = o;
}

// ---------------- WMMA bf16 GEMM, double-buffered, async global->LDS ----------------
// C[M,N] (+)= A[M,K] * B^T ; A: [M,K] row-major, B: [N,K] row-major (contiguous K)
// MODE 0: Cb = bf16(acc + bias?)        MODE 1: Cb = bf16(gelu(acc + bias))
// MODE 2: Cf += acc (+ bias?)           MODE 3: Cf  = acc*scale + bias?
template <int BM, int BN, int WM, int WN, int MODE>
__global__ __launch_bounds__((BM / WM) * (BN / WN) * 32, 1)
void sa_gemm_wmma(const unsigned short* __restrict__ A, long long sA, int lda,
                  const unsigned short* __restrict__ Bp, long long sB, int ldb,
                  float* __restrict__ Cf, unsigned short* __restrict__ Cb,
                  long long sC, int ldc,
                  const float* __restrict__ bias, float scale, int Kdim) {
  constexpr int BK = 32;
  constexpr int WROWS = BM / WM, WCOLS = BN / WN;
  constexpr int NWAVE = WROWS * WCOLS;
  constexpr int T = NWAVE * 32;
  constexpr int LDA = BK + 8;   // u16 units, rows 16B-aligned (80 bytes)
  constexpr int LDB = BK + 8;
  constexpr int ABUF = BM * LDA;
  constexpr int BBUF = BN * LDB;
  __shared__ unsigned short lds_a[2 * ABUF];
  __shared__ unsigned short lds_b[2 * BBUF];

  const int tid  = threadIdx.x;
  const int wave = tid >> 5;
  const int lane = tid & 31;
  const int lrow = lane & 15;
  const int half = lane >> 4;
  const int wm = wave % WROWS;
  const int wn = wave / WROWS;

  const int bm0 = blockIdx.y * BM;
  const int bn0 = blockIdx.x * BN;
  const int z   = blockIdx.z;
  A  += (long long)z * sA + (long long)bm0 * lda;   // tile row base folded in
  Bp += (long long)z * sB + (long long)bn0 * ldb;

  constexpr int MT = WM / 16, NT = WN / 16;
  v8f acc[MT][NT];
#pragma unroll
  for (int i = 0; i < MT; ++i)
#pragma unroll
    for (int j = 0; j < NT; ++j) acc[i][j] = v8f{0.f, 0.f, 0.f, 0.f, 0.f, 0.f, 0.f, 0.f};

  constexpr int AOCT = (BM * BK) / 8 / T;          // 16B copies per thread
  constexpr int BOCT = (BN * BK) / 8 / T;

#if SA_ASYNC
  auto stageA = [&](int kb, unsigned short* buf) {
#pragma unroll
    for (int i = 0; i < AOCT; ++i) {
      int o = tid + i * T, row = o >> 2, oc = o & 3;
      sa_cp16_async(A + (long long)row * lda + kb + oc * 8, &buf[row * LDA + oc * 8]);
    }
  };
  auto stageB = [&](int kb, unsigned short* buf) {
#pragma unroll
    for (int i = 0; i < BOCT; ++i) {
      int o = tid + i * T, row = o >> 2, oc = o & 3;
      sa_cp16_async(Bp + (long long)row * ldb + kb + oc * 8, &buf[row * LDB + oc * 8]);
    }
  };
#else
  uint4 ra[AOCT], rb[BOCT];
  auto loadA = [&](int kb) {
#pragma unroll
    for (int i = 0; i < AOCT; ++i) {
      int o = tid + i * T, row = o >> 2, oc = o & 3;
      ra[i] = *(const uint4*)(A + (long long)row * lda + kb + oc * 8);
    }
  };
  auto loadB = [&](int kb) {
#pragma unroll
    for (int i = 0; i < BOCT; ++i) {
      int o = tid + i * T, row = o >> 2, oc = o & 3;
      rb[i] = *(const uint4*)(Bp + (long long)row * ldb + kb + oc * 8);
    }
  };
  auto storeA = [&](unsigned short* buf) {
#pragma unroll
    for (int i = 0; i < AOCT; ++i) {
      int o = tid + i * T, row = o >> 2, oc = o & 3;
      *(uint4*)&buf[row * LDA + oc * 8] = ra[i];
    }
  };
  auto storeB = [&](unsigned short* buf) {
#pragma unroll
    for (int i = 0; i < BOCT; ++i) {
      int o = tid + i * T, row = o >> 2, oc = o & 3;
      *(uint4*)&buf[row * LDB + oc * 8] = rb[i];
    }
  };
#endif

  // -------- prologue: tile 0 --------
  const int nt = Kdim / BK;
#if SA_ASYNC
  stageA(0, lds_a); stageB(0, lds_b);
  sa_async_wait0();
#else
  loadA(0); loadB(0);
  storeA(lds_a); storeB(lds_b);
#endif
  __syncthreads();

  for (int t = 0; t < nt; ++t) {
    unsigned short* curA = lds_a + (t & 1) * ABUF;
    unsigned short* curB = lds_b + (t & 1) * BBUF;
    const bool more = (t + 1 < nt);
#if SA_ASYNC
    // next-tile copies go straight to the free LDS buffer; no data VGPRs.
    if (more) {
      stageA((t + 1) * BK, lds_a + ((t + 1) & 1) * ABUF);
      stageB((t + 1) * BK, lds_b + ((t + 1) & 1) * BBUF);
    }
#else
    if (more) { loadA((t + 1) * BK); loadB((t + 1) * BK); }
#endif

    // ---- fragments (ISA 16-bit A/B layout: bf16 pairs per VGPR) ----
    Frag16 fa[MT], fb[NT];
#pragma unroll
    for (int i = 0; i < MT; ++i) {
      int row = wm * WM + i * 16 + lrow;
#pragma unroll
      for (int v = 0; v < 8; ++v) {
        int koff = ((v >> 2) << 4) + (half << 3) + ((v & 3) << 1);
        fa[i].u[v] = *(const unsigned int*)&curA[row * LDA + koff];
      }
    }
#pragma unroll
    for (int j = 0; j < NT; ++j) {
      int row = wn * WN + j * 16 + lrow;
#pragma unroll
      for (int v = 0; v < 8; ++v) {
        int koff = ((v >> 2) << 4) + (half << 3) + ((v & 3) << 1);
        fb[j].u[v] = *(const unsigned int*)&curB[row * LDB + koff];
      }
    }
#pragma unroll
    for (int i = 0; i < MT; ++i)
#pragma unroll
      for (int j = 0; j < NT; ++j)
        acc[i][j] = __builtin_amdgcn_wmma_f32_16x16x32_bf16(
            false, fa[i].v, false, fb[j].v, (short)0, acc[i][j], false, false);

#if SA_ASYNC
    if (more) sa_async_wait0();
#else
    if (more) {
      storeA(lds_a + ((t + 1) & 1) * ABUF);
      storeB(lds_b + ((t + 1) & 1) * BBUF);
    }
#endif
    __syncthreads();
  }

  // -------- epilogue --------
#pragma unroll
  for (int i = 0; i < MT; ++i) {
#pragma unroll
    for (int j = 0; j < NT; ++j) {
      int n = bn0 + wn * WN + j * 16 + lrow;
      float bn = (MODE == 1) ? bias[n] : (bias ? bias[n] : 0.f);
#pragma unroll
      for (int r = 0; r < 8; ++r) {
        int m = bm0 + wm * WM + i * 16 + half * 8 + r;
        float val = acc[i][j][r];
        long long idx = (long long)z * sC + (long long)m * ldc + n;
        if constexpr (MODE == 0) {
          Cb[idx] = f2bf(val + bn);
        } else if constexpr (MODE == 1) {
          Cb[idx] = f2bf(gelu_tanh(val + bn));
        } else if constexpr (MODE == 2) {
          Cf[idx] += val + bn;
        } else {
          Cf[idx] = val * scale + bn;
        }
      }
    }
  }
}

// ---------------- host orchestration ----------------
extern "C" void kernel_launch(void* const* d_in, const int* in_sizes, int n_in,
                              void* d_out, int out_size, void* d_ws, size_t ws_size,
                              hipStream_t stream) {
  (void)in_sizes; (void)n_in; (void)out_size; (void)ws_size;
  const float* inputs    = (const float*)d_in[0];
  const float* slots_emb = (const float*)d_in[1];
  const float* ln_in_w = (const float*)d_in[2];
  const float* ln_in_b = (const float*)d_in[3];
  const float* ln_sl_w = (const float*)d_in[4];
  const float* ln_sl_b = (const float*)d_in[5];
  const float* ln_ff_w = (const float*)d_in[6];
  const float* ln_ff_b = (const float*)d_in[7];
  const float* Wq = (const float*)d_in[8];  const float* bq = (const float*)d_in[9];
  const float* Wk = (const float*)d_in[10]; const float* bk = (const float*)d_in[11];
  const float* Wv = (const float*)d_in[12]; const float* bv = (const float*)d_in[13];
  const float* W1 = (const float*)d_in[14]; const float* b1 = (const float*)d_in[15];
  const float* W2 = (const float*)d_in[16]; const float* b2 = (const float*)d_in[17];
  const float* Wh = (const float*)d_in[18]; const float* bh = (const float*)d_in[19];
  float* out = (float*)d_out;

  size_t off = 0;
  auto alloc = [&](size_t bytes) -> void* {
    void* p = (char*)d_ws + off;
    off += (bytes + 255) & ~(size_t)255;
    return p;
  };
  unsigned short* Xn  = (unsigned short*)alloc((size_t)B_ * N_ * D_ * 2);
  unsigned short* Kb  = (unsigned short*)alloc((size_t)B_ * N_ * D_ * 2);
  unsigned short* Vb  = (unsigned short*)alloc((size_t)B_ * N_ * D_ * 2);
  unsigned short* Vt  = (unsigned short*)alloc((size_t)B_ * N_ * D_ * 2);  // [b][D][N]
  unsigned short* Wqt = (unsigned short*)alloc((size_t)D_ * D_ * 2);       // [out][in]
  unsigned short* Wkt = (unsigned short*)alloc((size_t)D_ * D_ * 2);
  unsigned short* Wvt = (unsigned short*)alloc((size_t)D_ * D_ * 2);
  unsigned short* W1t = (unsigned short*)alloc((size_t)H_ * D_ * 2);       // [H][D]
  unsigned short* W2t = (unsigned short*)alloc((size_t)D_ * H_ * 2);       // [D][H]
  unsigned short* Wht = (unsigned short*)alloc((size_t)D_ * D_ * 2);
  float*          slots = (float*)alloc((size_t)B_ * S_ * D_ * 4);
  unsigned short* Sn  = (unsigned short*)alloc((size_t)B_ * S_ * D_ * 2);
  unsigned short* qb  = (unsigned short*)alloc((size_t)B_ * S_ * D_ * 2);
  float*          dots = (float*)alloc((size_t)B_ * S_ * N_ * 4);
  unsigned short* attnb = (unsigned short*)alloc((size_t)B_ * S_ * N_ * 2);
  unsigned short* hb  = (unsigned short*)alloc((size_t)B_ * S_ * H_ * 2);

  const float scale_dots = 0.03125f;   // D^-0.5 = 1/32

  // weights -> transposed bf16 ([out, in]; every GEMM B is then contiguous-K)
  sa_cvt_bf16_t<<<dim3(D_ / 32, D_ / 32), 256, 0, stream>>>(Wq, Wqt, D_, D_);
  sa_cvt_bf16_t<<<dim3(D_ / 32, D_ / 32), 256, 0, stream>>>(Wk, Wkt, D_, D_);
  sa_cvt_bf16_t<<<dim3(D_ / 32, D_ / 32), 256, 0, stream>>>(Wv, Wvt, D_, D_);
  sa_cvt_bf16_t<<<dim3(H_ / 32, D_ / 32), 256, 0, stream>>>(W1, W1t, D_, H_);
  sa_cvt_bf16_t<<<dim3(D_ / 32, H_ / 32), 256, 0, stream>>>(W2, W2t, H_, D_);
  sa_cvt_bf16_t<<<dim3(D_ / 32, D_ / 32), 256, 0, stream>>>(Wh, Wht, D_, D_);

  // x = LN(inputs)
  sa_ln_bf16<<<B_ * N_, 256, 0, stream>>>(inputs, ln_in_w, ln_in_b, Xn);

  // K = x@Wk + bk ; V = x@Wv + bv   (M=32768, N=1024, K=1024)
  dim3 gKV(N_ / 128, (B_ * N_) / 128, 1);
  sa_gemm_wmma<128, 128, 32, 64, 0><<<gKV, 256, 0, stream>>>(
      Xn, 0, D_, Wkt, 0, D_, nullptr, Kb, 0, D_, bk, 1.f, D_);
  sa_gemm_wmma<128, 128, 32, 64, 0><<<gKV, 256, 0, stream>>>(
      Xn, 0, D_, Wvt, 0, D_, nullptr, Vb, 0, D_, bv, 1.f, D_);

  // V^T per batch (invariant across iterations)
  sa_transpose_bf16<<<dim3(D_ / 32, N_ / 32, B_), 256, 0, stream>>>(Vb, Vt);

  // slots = broadcast(slots_emb)
  sa_init_slots<<<(B_ * S_ * D_ / 4) / 256, 256, 0, stream>>>(slots_emb, slots);

  dim3 gQ(D_ / 128, (B_ * S_) / 128, 1);        // (8,16)
  dim3 gDot(N_ / 128, 1, B_);                   // (8,1,32)
  dim3 gUpd(D_ / 128, 1, B_);                   // (8,1,32)
  dim3 gF1(H_ / 128, (B_ * S_) / 128, 1);       // (32,16)
  dim3 gF2(D_ / 128, (B_ * S_) / 128, 1);       // (8,16)

  for (int t = 0; t < 3; ++t) {
    // s = LN(slots); q = s@Wq + bq
    sa_ln_bf16<<<B_ * S_, 256, 0, stream>>>(slots, ln_sl_w, ln_sl_b, Sn);
    sa_gemm_wmma<128, 128, 32, 64, 0><<<gQ, 256, 0, stream>>>(
        Sn, 0, D_, Wqt, 0, D_, nullptr, qb, 0, D_, bq, 1.f, D_);

    // dots = scale * q @ K^T  (batched; K rows are [n, d], contraction over d)
    sa_gemm_wmma<64, 128, 16, 64, 3><<<gDot, 256, 0, stream>>>(
        qb, (long long)S_ * D_, D_, Kb, (long long)N_ * D_, D_,
        dots, nullptr, (long long)S_ * N_, N_, nullptr, scale_dots, D_);

    // softmax over slots, +eps, renorm over keys -> bf16 attn
    sa_softmax_slots<<<dim3(N_ / 256, B_), 256, 0, stream>>>(dots);
    sa_renorm<<<dim3(S_, B_), 256, 0, stream>>>(dots, attnb);

    // slots += attn @ V  (batched; B = V^T rows [d, keys], f32 accumulate)
    sa_gemm_wmma<64, 128, 16, 64, 2><<<gUpd, 256, 0, stream>>>(
        attnb, (long long)S_ * N_, N_, Vt, (long long)D_ * N_, N_,
        slots, nullptr, (long long)S_ * D_, D_, nullptr, 1.f, N_);

    // h = LN(slots); slots += gelu(h@W1+b1)@W2 + b2
    sa_ln_bf16<<<B_ * S_, 256, 0, stream>>>(slots, ln_ff_w, ln_ff_b, Sn);
    sa_gemm_wmma<128, 128, 32, 64, 1><<<gF1, 256, 0, stream>>>(
        Sn, 0, D_, W1t, 0, D_, nullptr, hb, 0, H_, b1, 1.f, D_);
    sa_gemm_wmma<128, 128, 32, 64, 2><<<gF2, 256, 0, stream>>>(
        hb, 0, H_, W2t, 0, H_, slots, nullptr, 0, D_, b2, 1.f, H_);
  }

  // out = slots @ Wh + bh  (slots converted to bf16 first)
  sa_cvt_bf16<<<(B_ * S_ * D_ / 4) / 256, 256, 0, stream>>>(slots, Sn);
  sa_gemm_wmma<128, 128, 32, 64, 3><<<gF2, 256, 0, stream>>>(
      Sn, 0, D_, Wht, 0, D_, out, nullptr, 0, D_, bh, 1.f, D_);
}
